// ScannedRNN_6914897346994
// MI455X (gfx1250) — matrix operand order repaired and checked by reading, not compiled, not run on hardware
//
#include <hip/hip_runtime.h>
#include <stdint.h>

// GRU scan for MI455X (gfx1250): persistent kernel, wave32, bf16 WMMA.
// T=1024, B=128, H=256. Grid = 16 WGs (one per 16-col group of H) x 8 waves
// (one per 16-row tile of B). Per step each wave computes six 16x16xK=256
// GEMM tiles (x-proj r/z/n and h-proj r/z/n) with v_wmma_f32_16x16x32_bf16,
// then the GRU gate math in fp32 (hardware v_tanh), then a device-wide barrier.

#define T_STEPS 1024
#define BATCH   128
#define HID     256
#define HID3    (3 * HID)
#define NWG     16
#define THREADS 256   // 8 waves of 32

typedef __bf16 v16bf __attribute__((ext_vector_type(16)));
typedef float  v8f   __attribute__((ext_vector_type(8)));

union BFrag { uint4 u[2]; v16bf v; };

// A-matrix 16x32 bf16 fragment: lane l holds row m=l&15;
// klo = 16*(l>>4); elems 0..15 -> k=kc+klo+i (matches B fragment).
__device__ __forceinline__ v16bf load_a_frag(const float* __restrict__ p, int klo, float scale) {
  float4 x0 = *(const float4*)(p + klo);
  float4 x1 = *(const float4*)(p + klo + 4);
  float4 x2 = *(const float4*)(p + klo + 8);
  float4 x3 = *(const float4*)(p + klo + 12);
  v16bf a;
  a[0]  = (__bf16)(x0.x * scale); a[1]  = (__bf16)(x0.y * scale);
  a[2]  = (__bf16)(x0.z * scale); a[3]  = (__bf16)(x0.w * scale);
  a[4]  = (__bf16)(x1.x * scale); a[5]  = (__bf16)(x1.y * scale);
  a[6]  = (__bf16)(x1.z * scale); a[7]  = (__bf16)(x1.w * scale);
  a[8]  = (__bf16)(x2.x * scale); a[9]  = (__bf16)(x2.y * scale);
  a[10] = (__bf16)(x2.z * scale); a[11] = (__bf16)(x2.w * scale);
  a[12] = (__bf16)(x3.x * scale); a[13] = (__bf16)(x3.y * scale);
  a[14] = (__bf16)(x3.z * scale); a[15] = (__bf16)(x3.w * scale);
  return a;
}

// B-matrix 32x16 bf16 fragment: lane l holds col n=l&15, k = kc + 16*(l>>4) + i.
// Weights staged in LDS transposed ([n][k], k contiguous) -> two ds_load_b128.
__device__ __forceinline__ v16bf load_b_frag(const __bf16* p) {
  BFrag b;
  b.u[0] = *(const uint4*)(p);
  b.u[1] = *(const uint4*)(p + 8);
  return b.v;
}

// ---- fast activations: avoid IEEE divide / libm on the serial critical path ----
__device__ __forceinline__ float fast_rcp(float x) {
#if __has_builtin(__builtin_amdgcn_rcpf)
  return __builtin_amdgcn_rcpf(x);         // v_rcp_f32
#else
  return 1.0f / x;
#endif
}

__device__ __forceinline__ float fast_tanh(float x) {
#if __has_builtin(__builtin_amdgcn_tanhf)
  return __builtin_amdgcn_tanhf(x);        // gfx1250 v_tanh_f32 (single TRANS op)
#else
  // tanh(x) = 1 - 2/(exp(2x)+1); saturates correctly at +/-1.
  return __builtin_fmaf(-2.0f, fast_rcp(__expf(2.0f * x) + 1.0f), 1.0f);
#endif
}

__device__ __forceinline__ float fast_sigmoid(float x) {
  // Exact identity: sigmoid(x) = 0.5*tanh(x/2) + 0.5
  return __builtin_fmaf(0.5f, fast_tanh(0.5f * x), 0.5f);
}

__global__ __launch_bounds__(THREADS)
void gru_scan_kernel(const float* __restrict__ ins,
                     const uint8_t* __restrict__ resets,
                     const float* __restrict__ Wi, const float* __restrict__ bi,
                     const float* __restrict__ Wh, const float* __restrict__ bhn,
                     const float* __restrict__ h0,
                     float* __restrict__ out_final, float* __restrict__ out_ys,
                     int* __restrict__ counters,
                     float* __restrict__ hbuf0, float* __restrict__ hbuf1) {
  // Resident bf16 weight slices, transposed: [gate][n(16 cols)][k(256)], 24KB each.
  __shared__ __bf16 WsI[3][16][HID];
  __shared__ __bf16 WsH[3][16][HID];

  const int g    = blockIdx.x;      // column group: owns H-cols [16g, 16g+16)
  const int tid  = threadIdx.x;
  const int wave = tid >> 5;
  const int lane = tid & 31;

  // ---- one-time: stage weights (fp32 -> bf16) into LDS ----
  for (int idx = tid; idx < 3 * 16 * HID; idx += THREADS) {
    int q = idx >> 12;          // gate 0..2
    int r = idx & 4095;
    int n = r >> 8;             // 0..15
    int k = r & 255;            // 0..255
    int col = q * HID + g * 16 + n;
    WsI[q][n][k] = (__bf16)Wi[k * HID3 + col];
    WsH[q][n][k] = (__bf16)Wh[k * HID3 + col];
  }
  // ---- one-time: init h double-buffer (this WG's columns) ----
  for (int idx = tid; idx < BATCH * 16; idx += THREADS) {
    int row = idx >> 4;
    int j   = g * 16 + (idx & 15);
    hbuf0[row * HID + j] = h0[row * HID + j];
  }

  // Per-lane biases for this lane's output column (lanes 16..31 mirror 0..15).
  const int   ncol  = lane & 15;
  const int   jcol  = g * 16 + ncol;
  const float bi_r  = bi[jcol];
  const float bi_z  = bi[HID + jcol];
  const float bi_n  = bi[2 * HID + jcol];
  const float bhn_v = bhn[jcol];

  const int m0   = wave * 16;             // row-tile base (B rows)
  const int arow = m0 + (lane & 15);      // A-fragment row for this lane
  const int klo  = (lane >> 4) * 16;      // A-fragment k sub-offset
  const int kbB  = (lane >> 4) * 16;      // B-fragment k sub-offset
  const int rgrp = (lane >> 4) * 8;       // C/D row offset for this half-wave
  const int grow = m0 + rgrp;             // first gate-math row for this lane

  // ---- device-wide barrier via per-step counters (zeroed by host memset) ----
  auto grid_barrier = [&](int slot) {
    __syncthreads();
    __threadfence();                       // release stores device-wide
    if (tid == 0) {
      __hip_atomic_fetch_add(&counters[slot], 1, __ATOMIC_RELEASE, __HIP_MEMORY_SCOPE_AGENT);
      while (__hip_atomic_load(&counters[slot], __ATOMIC_ACQUIRE, __HIP_MEMORY_SCOPE_AGENT) < NWG)
        __builtin_amdgcn_s_sleep(1);
    }
    __syncthreads();
    __threadfence();                       // acquire: invalidate stale near caches
  };

  grid_barrier(0);                         // weights + h0 visible everywhere

  for (int t = 0; t < T_STEPS; ++t) {
    const float* hcur = (t & 1) ? hbuf1 : hbuf0;
    float*       hnxt = (t & 1) ? hbuf0 : hbuf1;
    const float* xrow = ins + ((size_t)t * BATCH + arow) * HID;
    const float* hrow = hcur + (size_t)arow * HID;
    const float  rmask = resets[t * BATCH + arow] ? 0.0f : 1.0f;
    const float* xpref = (t + 1 < T_STEPS) ? (xrow + (size_t)BATCH * HID) : xrow;

    // Gate-row reset flags: 8 consecutive bytes, 8B-aligned -> one load.
    const uint2 fl = *(const uint2*)(resets + t * BATCH + grow);
    // Carry values for the z*h term: unconditional strided loads, issued early
    // so their latency overlaps the GEMM below.
    float hv[8];
#pragma unroll
    for (int r = 0; r < 8; ++r) hv[r] = hcur[(grow + r) * HID + jcol];

    v8f accX[3] = {};   // xr, xz, xn tiles (f32 accum)
    v8f accH[3] = {};   // hr, hz, hn tiles

    for (int kc = 0; kc < HID; kc += 32) {
      __builtin_prefetch(xpref + kc + klo, 0, 1);          // next-step ins rows
      v16bf ax = load_a_frag(xrow + kc, klo, 1.0f);
      v16bf ah = load_a_frag(hrow + kc, klo, rmask);       // reset applied to carry
#pragma unroll
      for (int q = 0; q < 3; ++q) {
        v16bf bI = load_b_frag(&WsI[q][ncol][kc + kbB]);
        v16bf bH = load_b_frag(&WsH[q][ncol][kc + kbB]);
        accX[q] = __builtin_amdgcn_wmma_f32_16x16x32_bf16(
            false, ax, false, bI, (short)0, accX[q], false, false);
        accH[q] = __builtin_amdgcn_wmma_f32_16x16x32_bf16(
            false, ah, false, bH, (short)0, accH[q], false, false);
      }
    }

    // ---- GRU gate math (fp32), C/D layout: VGPR r -> row grow+r, N = ncol ----
#pragma unroll
    for (int r = 0; r < 8; ++r) {
      const int      row  = grow + r;
      const uint32_t fb   = (r < 4) ? ((fl.x >> (8 * r)) & 0xffu)
                                    : ((fl.y >> (8 * (r - 4))) & 0xffu);
      const float    hm   = hv[r] * (fb ? 0.0f : 1.0f);    // branch-free reset
      const float    rr   = fast_sigmoid(accX[0][r] + bi_r + accH[0][r]);
      const float    zz   = fast_sigmoid(accX[1][r] + bi_z + accH[1][r]);
      const float    nn   = fast_tanh(accX[2][r] + bi_n + rr * (accH[2][r] + bhn_v));
      const float    nh   = (1.0f - zz) * nn + zz * hm;
      hnxt[row * HID + jcol] = nh;
      __builtin_nontemporal_store(nh, &out_ys[((size_t)t * BATCH + row) * HID + jcol]);
    }

    grid_barrier(t + 1);
  }

  // Final carry: after an even number of steps it lives in hbuf0, written by
  // this very lane -> no barrier or branch-in-loop needed.
  const float* hfin = (T_STEPS & 1) ? hbuf1 : hbuf0;
#pragma unroll
  for (int r = 0; r < 8; ++r) {
    const int row = grow + r;
    out_final[row * HID + jcol] = hfin[row * HID + jcol];
  }
}

extern "C" void kernel_launch(void* const* d_in, const int* in_sizes, int n_in,
                              void* d_out, int out_size, void* d_ws, size_t ws_size,
                              hipStream_t stream) {
  (void)in_sizes; (void)n_in; (void)out_size; (void)ws_size;
  const float*   ins    = (const float*)d_in[0];
  const uint8_t* resets = (const uint8_t*)d_in[1];   // jnp.bool_ -> 1 byte/elem
  const float*   Wi     = (const float*)d_in[2];
  const float*   bi     = (const float*)d_in[3];
  const float*   Wh     = (const float*)d_in[4];
  const float*   bhn    = (const float*)d_in[5];
  const float*   h0     = (const float*)d_in[6];

  float* out       = (float*)d_out;
  float* out_final = out;                       // [B,H] first (return order)
  float* out_ys    = out + BATCH * HID;         // then [T,B,H]

  // Workspace: counters (T+1 ints) at 0, then h double-buffer. ~270 KB total.
  int*   counters = (int*)d_ws;
  float* hbuf0    = (float*)((char*)d_ws + 8192);
  float* hbuf1    = hbuf0 + BATCH * HID;

  // Graph-capture-safe per-launch zeroing of the barrier counters.
  hipMemsetAsync(counters, 0, (T_STEPS + 1) * sizeof(int), stream);

  gru_scan_kernel<<<NWG, THREADS, 0, stream>>>(ins, resets, Wi, bi, Wh, bhn, h0,
                                               out_final, out_ys,
                                               counters, hbuf0, hbuf1);
}
